// SDE_28166395527951
// MI455X (gfx1250) — compile-verified, matrix operand order checked by reading, not run on hardware
//
#include <hip/hip_runtime.h>
#include <hip/hip_bf16.h>

// ---------------------------------------------------------------------------
// SDE integrator on gfx1250 (MI455X).
//
//   * Batch on the WMMA N dimension (16 cols/tile, 2 tiles/wave).
//   * Hidden 64x64 layers: D = A x B + C, A = W^T fragments from LDS,
//     B = activations, C = pre-splatted per-lane bias fragment from LDS.
//   * D layout feeds the next layer's B operand lane-locally:
//       B.slot[s] of chunk c <-> K-feature (s&7) + 16*(s>=8) + 32c + 8*laneHi
//       D tile t, VGPR v     <-> feature   v + 8*laneHi + 16t
//     chunk0 = pack(D_t0,D_t1), chunk1 = pack(D_t2,D_t3); relu done in f16
//     (single v_pk_max_num_f16) after the f32->f16 convert.
//   * First layers (1->64 / 2->64): rank-1/2 packed-f16 VALU against
//     pre-swizzled vectors in B-slot order.
//   * Last layers (64->1): per-lane f32 dot over relu(D) + ds_swizzle SWAPX16.
//   * asm memory fences keep step-invariant LDS loads inside the step loop
//     (otherwise LICM hoists ~100KB into registers and spills to scratch).
// ---------------------------------------------------------------------------

typedef __attribute__((ext_vector_type(16))) _Float16 v16h;
typedef __attribute__((ext_vector_type(8)))  float    v8f;
typedef __attribute__((ext_vector_type(4)))  float    v4f;
typedef __attribute__((ext_vector_type(2)))  float    v2f;

#define MEM_FENCE() asm volatile("" ::: "memory")

// Packed-blob layout (bytes) in d_ws, mirrored verbatim into LDS.
constexpr int OFF_AFRAG  = 0;        // 10 layers * 8 frags * 1024B      = 81920
constexpr int OFF_CBIAS  = 81920;    // 10 layers * 4 tiles * 1024B      = 40960
constexpr int OFF_FVEC   = 122880;   // 6 rows * 2 chunks * 1024B        = 12288
constexpr int OFF_BVEC   = 135168;   // 4 mlps * 2 chunks * 1024B        =  8192
constexpr int OFF_LASTW  = 143360;   // 4 * 64 f32                       =  1024
constexpr int OFF_LASTB  = 144384;   // 4 f32
constexpr int SMEM_BYTES = 144512;   // 16B multiple

constexpr int NSTEPS = 100;
constexpr int BATCH  = 262144;

// ---------------------------------------------------------------------------

__device__ __forceinline__ v8f wmma_f16(v16h a, v16h b, v8f c) {
  return __builtin_amdgcn_wmma_f32_16x16x32_f16(
      /*neg_a=*/false, a, /*neg_b=*/false, b,
      /*c_mod=*/(short)0, c, /*reuse_a=*/false, /*reuse_b=*/false);
}

// Convert then relu in f16: one v_cvt per element + one pk_max per pair.
__device__ __forceinline__ v16h pack_relu(v8f d0, v8f d1) {
  v16h r;
#pragma unroll
  for (int i = 0; i < 8; ++i) {
    r[i]     = (_Float16)d0[i];
    r[i + 8] = (_Float16)d1[i];
  }
#pragma unroll
  for (int i = 0; i < 16; ++i) r[i] = __builtin_fmaxf16(r[i], (_Float16)0.f);
  return r;
}

// L0: first hidden-layer index (0..9), NHID: #64x64 layers, TWOIN: 2-wide input,
// ROW0: first-layer weight-row table index, MLP: mlp index (bias/last tables).
template <int L0, int NHID, bool TWOIN, int ROW0, int MLP>
__device__ __forceinline__ void run_mlp(const char* sm, int lane,
                                        _Float16 x0t0, _Float16 x0t1,
                                        _Float16 x1t0, _Float16 x1t1,
                                        float& out0, float& out1) {
  const int laneHi = lane >> 4;
  const _Float16 xs0[2] = {x0t0, x0t1};
  const _Float16 xs1[2] = {x1t0, x1t1};

  // ---- first layer (rank-1/2) straight into B-operand layout -------------
  MEM_FENCE();
  v16h Bc[2][2];
#pragma unroll
  for (int c = 0; c < 2; ++c) {
    v16h w0 = *(const v16h*)(sm + OFF_FVEC + (ROW0 * 2 + c) * 1024 + lane * 32);
    v16h bv = *(const v16h*)(sm + OFF_BVEC + (MLP * 2 + c) * 1024 + lane * 32);
    v16h w1 = {};
    if constexpr (TWOIN)
      w1 = *(const v16h*)(sm + OFF_FVEC + ((ROW0 + 1) * 2 + c) * 1024 + lane * 32);
#pragma unroll
    for (int tl = 0; tl < 2; ++tl) {
      v16h r;
#pragma unroll
      for (int i = 0; i < 16; ++i) {
        _Float16 a = bv[i] + xs0[tl] * w0[i];
        if constexpr (TWOIN) a = a + xs1[tl] * w1[i];
        r[i] = __builtin_fmaxf16(a, (_Float16)0.f);
      }
      Bc[tl][c] = r;
    }
  }

  // ---- hidden 64x64 layers via WMMA --------------------------------------
  v8f D[2][4];
#pragma unroll
  for (int L = 0; L < NHID; ++L) {
    MEM_FENCE();  // keep this layer's fragment loads local to this layer
#pragma unroll
    for (int t = 0; t < 4; ++t) {
      const char* fb = sm + OFF_AFRAG + (L0 + L) * 8192 + (t * 2) * 1024 + lane * 32;
      v16h A0 = *(const v16h*)fb;            // K = 0..31 fragment of W^T
      v16h A1 = *(const v16h*)(fb + 1024);   // K = 32..63 fragment
      // pre-splatted bias fragment (per-lane, D layout) seeds C
      v8f cseed = *(const v8f*)(sm + OFF_CBIAS + ((L0 + L) * 4 + t) * 1024 + lane * 32);
#pragma unroll
      for (int tl = 0; tl < 2; ++tl) {
        v8f acc = wmma_f16(A0, Bc[tl][0], cseed);
        acc = wmma_f16(A1, Bc[tl][1], acc);
        D[tl][t] = acc;
      }
    }
    if (L < NHID - 1) {
#pragma unroll
      for (int tl = 0; tl < 2; ++tl) {
        Bc[tl][0] = pack_relu(D[tl][0], D[tl][1]);
        Bc[tl][1] = pack_relu(D[tl][2], D[tl][3]);
      }
    }
  }

  // ---- last layer (64 -> 1): per-lane dot over relu(D), then SWAPX16 -----
  MEM_FENCE();
  float ps0 = 0.f, ps1 = 0.f;
#pragma unroll
  for (int t = 0; t < 4; ++t) {
    const v4f* wv = (const v4f*)(sm + OFF_LASTW + MLP * 256 + (16 * t + 8 * laneHi) * 4);
    v4f wa = wv[0], wb = wv[1];
#pragma unroll
    for (int i = 0; i < 4; ++i) {
      float d00 = __builtin_fmaxf(D[0][t][i],     0.f);
      float d01 = __builtin_fmaxf(D[0][t][i + 4], 0.f);
      float d10 = __builtin_fmaxf(D[1][t][i],     0.f);
      float d11 = __builtin_fmaxf(D[1][t][i + 4], 0.f);
      ps0 += d00 * wa[i] + d01 * wb[i];
      ps1 += d10 * wa[i] + d11 * wb[i];
    }
  }
  float lb = *(const float*)(sm + OFF_LASTB + MLP * 4);
  // lane l and l^16 hold complementary feature halves of the same column
  ps0 += __int_as_float(__builtin_amdgcn_ds_swizzle(__float_as_int(ps0), 0x401F));
  ps1 += __int_as_float(__builtin_amdgcn_ds_swizzle(__float_as_int(ps1), 0x401F));
  out0 = ps0 + lb;
  out1 = ps1 + lb;
}

// ---------------------------------------------------------------------------
// Prep kernel: repack all f32 weights into the lane-packed f16/f32 blob.
// ---------------------------------------------------------------------------
struct Params { const float* p[40]; };

__global__ void sde_prep(Params prm, char* blob) {
  const int HW[10] = {6, 8, 10, 16, 18, 20, 26, 28, 34, 36};   // hidden W idx
  const int HB[10] = {7, 9, 11, 17, 19, 21, 27, 29, 35, 37};   // hidden b idx
  const int FWP[6] = {4, 14, 14, 24, 32, 32};                  // first W param
  const int FWR[6] = {0, 0, 1, 0, 0, 1};                       // row within W1
  const int B1P[4] = {5, 15, 25, 33};                          // first b param
  const int LWP[4] = {12, 22, 30, 38};                         // last W param
  const int LBP[4] = {13, 23, 31, 39};                         // last b param

  int gid = blockIdx.x * blockDim.x + threadIdx.x;
  int nth = gridDim.x * blockDim.x;

  // A fragments: A = W^T, lane -> output feature (lane&15)+16t,
  // slot s -> input feature 32c + (s&7) + 16*(s>=8) + 8*laneHi.
  for (int idx = gid; idx < 10 * 8 * 32 * 16; idx += nth) {
    int s = idx & 15, lane = (idx >> 4) & 31, f = (idx >> 9) & 7, L = idx >> 12;
    int t = f >> 1, c = f & 1, laneHi = lane >> 4;
    int n = (lane & 15) + 16 * t;
    int k = 32 * c + (s & 7) + 16 * (s >> 3) + 8 * laneHi;
    float w = prm.p[HW[L]][k * 64 + n];
    ((_Float16*)(blob + OFF_AFRAG + L * 8192 + f * 1024 + lane * 32))[s] = (_Float16)w;
  }
  // Pre-splatted C bias fragments: 8 identical f32 per lane, D layout.
  for (int idx = gid; idx < 10 * 4 * 32 * 8; idx += nth) {
    int j = idx & 7, lane = (idx >> 3) & 31, t = (idx >> 8) & 3, L = idx >> 10;
    float b = prm.p[HB[L]][(lane & 15) + 16 * t];
    *(float*)(blob + OFF_CBIAS + (L * 4 + t) * 1024 + lane * 32 + j * 4) = b;
  }
  // First-layer weight vectors in B-slot order over output features.
  for (int idx = gid; idx < 6 * 2 * 32 * 16; idx += nth) {
    int s = idx & 15, lane = (idx >> 4) & 31, c = (idx >> 9) & 1, r = idx >> 10;
    int laneHi = lane >> 4;
    int fo = 32 * c + (s & 7) + 16 * (s >> 3) + 8 * laneHi;
    float w = prm.p[FWP[r]][FWR[r] * 64 + fo];
    ((_Float16*)(blob + OFF_FVEC + (r * 2 + c) * 1024 + lane * 32))[s] = (_Float16)w;
  }
  // First-layer bias vectors, same slot order.
  for (int idx = gid; idx < 4 * 2 * 32 * 16; idx += nth) {
    int s = idx & 15, lane = (idx >> 4) & 31, c = (idx >> 9) & 1, m = idx >> 10;
    int laneHi = lane >> 4;
    int fo = 32 * c + (s & 7) + 16 * (s >> 3) + 8 * laneHi;
    float b = prm.p[B1P[m]][fo];
    ((_Float16*)(blob + OFF_BVEC + (m * 2 + c) * 1024 + lane * 32))[s] = (_Float16)b;
  }
  // Raw last-layer weights (f32[64] per mlp).
  for (int idx = gid; idx < 4 * 64; idx += nth) {
    int m = idx >> 6, j = idx & 63;
    *(float*)(blob + OFF_LASTW + m * 256 + j * 4) = prm.p[LWP[m]][j];
  }
  if (gid < 4) *(float*)(blob + OFF_LASTB + gid * 4) = prm.p[LBP[gid]][0];
}

// ---------------------------------------------------------------------------
// Main kernel: one wave integrates 32 batch columns for all 100 steps.
// ---------------------------------------------------------------------------
__global__ __launch_bounds__(256) void sde_main(const float* __restrict__ latents,
                                                const float* __restrict__ dtv,
                                                const float* __restrict__ noise,
                                                const char* __restrict__ blob,
                                                float* __restrict__ out) {
  extern __shared__ char sm[];
  const int tid = threadIdx.x;
  for (int i = tid * 16; i < SMEM_BYTES; i += 256 * 16)
    *(v4f*)(sm + i) = *(const v4f*)(blob + i);
  __syncthreads();

  const int lane = tid & 31;
  const int wave = tid >> 5;
  const long wgid = (long)blockIdx.x * 8 + wave;
  const int col0 = (int)(wgid * 32) + (lane & 15);
  const int col1 = col0 + 16;

  v2f y[2];
  y[0] = *(const v2f*)(latents + (size_t)col0 * 2);
  y[1] = *(const v2f*)(latents + (size_t)col1 * 2);
  float hh[2], sq[2];
  hh[0] = dtv[col0] * (1.0f / NSTEPS); sq[0] = sqrtf(hh[0]);
  hh[1] = dtv[col1] * (1.0f / NSTEPS); sq[1] = sqrtf(hh[1]);

  const float* zp0 = noise + (size_t)col0 * 2;
  const float* zp1 = noise + (size_t)col1 * 2;

#pragma unroll 1
  for (int s = 0; s < NSTEPS; ++s) {
    MEM_FENCE();  // nothing memory-related may be hoisted out of the loop
    v2f z0v = *(const v2f*)zp0;
    v2f z1v = *(const v2f*)zp1;
    zp0 += (size_t)BATCH * 2;
    zp1 += (size_t)BATCH * 2;

    _Float16 x0t0 = (_Float16)y[0][0], x0t1 = (_Float16)y[1][0];
    _Float16 x1t0 = (_Float16)y[0][1], x1t1 = (_Float16)y[1][1];

    float muo0, muo1, mu2o0, mu2o1, sgo0, sgo1, sg2o0, sg2o1;
    run_mlp<0, 3, false, 0, 0>(sm, lane, x0t0, x0t1, x1t0, x1t1, muo0, muo1);
    run_mlp<3, 3, true,  1, 1>(sm, lane, x0t0, x0t1, x1t0, x1t1, mu2o0, mu2o1);
    run_mlp<6, 2, false, 3, 2>(sm, lane, x0t0, x0t1, x1t0, x1t1, sgo0, sgo1);
    run_mlp<8, 2, true,  4, 3>(sm, lane, x0t0, x0t1, x1t0, x1t1, sg2o0, sg2o1);

    y[0][0] += muo0  * hh[0] + sgo0  * sq[0] * z0v[0];
    y[0][1] += mu2o0 * hh[0] + sg2o0 * sq[0] * z0v[1];
    y[1][0] += muo1  * hh[1] + sgo1  * sq[1] * z1v[0];
    y[1][1] += mu2o1 * hh[1] + sg2o1 * sq[1] * z1v[1];
  }

  if (lane < 16) {  // laneHi duplicates laneLo state; store once
    *(v2f*)(out + (size_t)col0 * 2) = y[0];
    *(v2f*)(out + (size_t)col1 * 2) = y[1];
  }
}

// ---------------------------------------------------------------------------
extern "C" void kernel_launch(void* const* d_in, const int* in_sizes, int n_in,
                              void* d_out, int out_size, void* d_ws, size_t ws_size,
                              hipStream_t stream) {
  Params prm;
  for (int i = 0; i < 40 && i < n_in; ++i) prm.p[i] = (const float*)d_in[i];
  char* blob = (char*)d_ws;

  sde_prep<<<64, 256, 0, stream>>>(prm, blob);

  const float* latents = (const float*)d_in[0];
  const float* dtv     = (const float*)d_in[2];
  const float* noise   = (const float*)d_in[3];
  sde_main<<<1024, 256, SMEM_BYTES, stream>>>(latents, dtv, noise, blob,
                                              (float*)d_out);
}
